// GNN_Location_73272142070169
// MI455X (gfx1250) — matrix operand order repaired and checked by reading, not compiled, not run on hardware
//
#include <hip/hip_runtime.h>

// ---------------------------------------------------------------------------
// GNN location pipeline for MI455X (gfx1250, wave32, WMMA).
// Edge MLP (hot path, ~80 GFLOP + ~1.2B f32 atomics) runs on
// v_wmma_f32_16x16x32_bf16 with fp32 accumulate; node MLPs are fp32 VALU with
// LDS-cached weights; scatter-mean via global_atomic_add_f32 into L2-resident
// accumulators.
//
// Input order (setup_inputs dict order):
//   0 x[N,18] f32, 1 mask[N,18] f32, 2 A_in_pick[2,E] i32, 3 A_in_src[2,E] i32,
//   4 A_src_in_product[2,N] i32, 5 A_sta_in_product[2,N] i32,
//   6 A_src_in_sta[2,N] i32 (row0=sta_idx,row1=src_idx),
//   7 locs_cart[200,3] f32, 8 srcs_cart[500,3] f32,
//   9.. params leaves in JAX tree (sorted-key) order:
//   bip1(9) bip2(9) da1..da5(24 each) embed(5) proj(5) proj_c(5) proj_t(5)
// ---------------------------------------------------------------------------

#define NPROD 100000
#define NEDGE 2000000
#define NSTA  200
#define NSRC  500
#define INV_SCALE (1.0f/30000.0f)   // pos/1000/30 == pos/30000 (DA and BIP)

typedef __bf16 bf16_t;
typedef __attribute__((ext_vector_type(16))) __bf16       v16bf;
typedef __attribute__((ext_vector_type(8)))  float        v8f;
typedef __attribute__((ext_vector_type(4)))  unsigned int u32x4;
typedef __attribute__((ext_vector_type(4)))  float        f32x4;

__device__ __forceinline__ float prelu_f(float x, float a) { return x >= 0.0f ? x : a * x; }

__device__ __forceinline__ unsigned short f2bf_u(float f) {
  unsigned u = __builtin_bit_cast(unsigned, f);
  unsigned r = u + 0x7FFFu + ((u >> 16) & 1u);   // round-to-nearest-even
  return (unsigned short)(r >> 16);
}
__device__ __forceinline__ bf16_t f2bf(float f) {
  unsigned short s = f2bf_u(f);
  return __builtin_bit_cast(bf16_t, s);
}

// ---------------------------------------------------------------------------
// Edge propagation: msg = prelu(W[33,30]^T . concat(feat30, rel3) + b, a_merge)
// scattered (sum + count) into accum[tgt]. 16 edges per wave-tile.
//   A (loop-invariant) = W^T tile: lane holds channel row c=lane%16 (tile0)
//     and 16+lane%16 (tile1); K elems contiguous: K = 16*half + j.
//   B = bf16 features of the lane's edge: element j <-> K = 16*half + j;
//     rows K=30,31 overridden with rel.x, rel.y (A rows 30,31 = W rel rows).
//   D: lane owns edge col N=lane%16, channel rows M = 8*half + r.
// ---------------------------------------------------------------------------
__global__ void __launch_bounds__(256)
k_edge_wmma(const unsigned short* __restrict__ feat,   // [N,32] bf16 bits
            const int* __restrict__ e0, const int* __restrict__ e1,
            const f32x4* __restrict__ pos,             // [N] raw xyz
            const float* __restrict__ Wm,              // merge.w [33,30]
            const float* __restrict__ Bm,              // merge.b [30]
            const float* __restrict__ amp,             // a_merge scalar
            float* __restrict__ accum,                 // [N,32]
            float* __restrict__ cnt,                   // [N]
            int ntiles) {
  const int lane = threadIdx.x & 31;
  const int lm   = lane & 15;
  const int hf   = lane >> 4;            // 0: K 0..15, 1: K 16..31
  const float am = amp[0];

  // --- A fragments (weights), loaded once per wave -------------------------
  const int klo = hf ? 16 : 0;
  v16bf a0, a1;
#pragma unroll
  for (int j = 0; j < 16; ++j) {
    int k = klo + j;                                 // contiguous: K = 16*hf + j
    a0[j] = f2bf(Wm[k * 30 + lm]);                   // channel c = lm
    int c1 = 16 + lm;
    a1[j] = (c1 < 30) ? f2bf(Wm[k * 30 + c1]) : f2bf(0.0f);
  }
  // --- epilogue constants (rel_z weights + bias for owned channels) --------
  const int r0 = hf * 8;                             // D rows M = r0 + r
  float wz0[8], wz1[8], bb0[8], bb1[8];
#pragma unroll
  for (int r = 0; r < 8; ++r) {
    int c0 = r0 + r, c1 = 16 + r0 + r;
    wz0[r] = Wm[32 * 30 + c0];
    bb0[r] = Bm[c0];
    wz1[r] = (c1 < 30) ? Wm[32 * 30 + c1] : 0.0f;
    bb1[r] = (c1 < 30) ? Bm[c1] : 0.0f;
  }

  const int wave   = blockIdx.x * (blockDim.x >> 5) + (threadIdx.x >> 5);
  const int nwaves = gridDim.x * (blockDim.x >> 5);

  for (int t = wave; t < ntiles; t += nwaves) {
    const int e  = t * 16 + lm;                 // lane's edge (lanes l,l+16 share)
    __builtin_prefetch(e0 + (t + nwaves) * 16, 0, 1);   // global_prefetch_b8
    const int i0 = e0[e];
    const int i1 = e1[e];
    const f32x4 p0 = pos[i0], p1 = pos[i1];
    const float relx = (p0.x - p1.x) * INV_SCALE;
    const float rely = (p0.y - p1.y) * INV_SCALE;
    const float relz = (p0.z - p1.z) * INV_SCALE;

    // B fragment: 16 contiguous bf16 of feat[i0] starting at K = 16*hf
    const u32x4* fp = (const u32x4*)(feat + (size_t)i0 * 32 + hf * 16);
    union { u32x4 u[2]; v16bf v; } bu;
    bu.u[0] = fp[0];
    bu.u[1] = fp[1];
    v16bf b = bu.v;
    if (hf) {                                    // K=30,31 -> rel.x, rel.y
      b[14] = f2bf(relx);
      b[15] = f2bf(rely);
    }

    v8f c0 = {0.f, 0.f, 0.f, 0.f, 0.f, 0.f, 0.f, 0.f};
    v8f c1 = {0.f, 0.f, 0.f, 0.f, 0.f, 0.f, 0.f, 0.f};
    c0 = __builtin_amdgcn_wmma_f32_16x16x32_bf16(false, a0, false, b, (short)0, c0, false, false);
    c1 = __builtin_amdgcn_wmma_f32_16x16x32_bf16(false, a1, false, b, (short)0, c1, false, false);

    float* arow = accum + (size_t)i1 * 32;
#pragma unroll
    for (int r = 0; r < 8; ++r) {
      float v = c0[r] + relz * wz0[r] + bb0[r];
      v = v >= 0.0f ? v : am * v;
      atomicAdd(arow + (r0 + r), v);
      int ch1 = 16 + r0 + r;
      if (ch1 < 30) {
        float v1 = c1[r] + relz * wz1[r] + bb1[r];
        v1 = v1 >= 0.0f ? v1 : am * v1;
        atomicAdd(arow + ch1, v1);
      }
    }
    if (!hf) atomicAdd(cnt + i1, 1.0f);
  }
}

// ---------------------------------------------------------------------------
// Node-level kernels (fp32 VALU, weights in LDS)
// ---------------------------------------------------------------------------
__global__ void k_nodepos(const int* __restrict__ sta_idx, const int* __restrict__ src_idx,
                          const float* __restrict__ locs, const float* __restrict__ srcs,
                          f32x4* __restrict__ posS, f32x4* __restrict__ posR) {
  int n = blockIdx.x * blockDim.x + threadIdx.x;
  if (n >= NPROD) return;
  int s = sta_idx[n], r = src_idx[n];
  f32x4 a; a.x = locs[s*3]; a.y = locs[s*3+1]; a.z = locs[s*3+2]; a.w = 0.f;
  f32x4 b; b.x = srcs[r*3]; b.y = srcs[r*3+1]; b.z = srcs[r*3+2]; b.w = 0.f;
  posS[n] = a; posR[n] = b;
}

__global__ void k_embed(const float* __restrict__ mask,
                        const float* __restrict__ W0, const float* __restrict__ B0,
                        const float* __restrict__ Ap,
                        const float* __restrict__ W1, const float* __restrict__ B1,
                        float* __restrict__ mout) {
  __shared__ float sW0[18*20], sB0[20], sW1[20*10], sB1[10];
  for (int i = threadIdx.x; i < 18*20; i += blockDim.x) sW0[i] = W0[i];
  for (int i = threadIdx.x; i < 20;    i += blockDim.x) sB0[i] = B0[i];
  for (int i = threadIdx.x; i < 20*10; i += blockDim.x) sW1[i] = W1[i];
  for (int i = threadIdx.x; i < 10;    i += blockDim.x) sB1[i] = B1[i];
  __syncthreads();
  int n = blockIdx.x * blockDim.x + threadIdx.x;
  if (n >= NPROD) return;
  float a = Ap[0];
  float in[18], h[20];
  for (int k = 0; k < 18; ++k) in[k] = mask[n*18 + k];
  for (int j = 0; j < 20; ++j) {
    float s = sB0[j];
    for (int k = 0; k < 18; ++k) s += in[k] * sW0[k*20 + j];
    h[j] = prelu_f(s, a);
  }
  for (int j = 0; j < 10; ++j) {
    float s = sB1[j];
    for (int k = 0; k < 20; ++k) s += h[k] * sW1[k*10 + j];
    mout[n*10 + j] = s;
  }
}

// t0 = prelu(init(concat(h,m)),a0); featA/B = bf16(prelu(t0,a11/a12))
__global__ void k_init(const float* __restrict__ hin, int hstride, int din,
                       const float* __restrict__ m,
                       const float* __restrict__ W, const float* __restrict__ B,
                       const float* __restrict__ a0p, const float* __restrict__ a11p,
                       const float* __restrict__ a12p,
                       float* __restrict__ t0,
                       unsigned short* __restrict__ fA, unsigned short* __restrict__ fB) {
  __shared__ float sW[40*30], sB[30];
  int dtot = din + 10;
  for (int i = threadIdx.x; i < dtot*30; i += blockDim.x) sW[i] = W[i];
  for (int i = threadIdx.x; i < 30;      i += blockDim.x) sB[i] = B[i];
  __syncthreads();
  int n = blockIdx.x * blockDim.x + threadIdx.x;
  if (n >= NPROD) return;
  float a0 = a0p[0], a11 = a11p[0], a12 = a12p[0];
  float in[40];
  for (int k = 0; k < din; ++k) in[k] = hin[(size_t)n*hstride + k];
  for (int k = 0; k < 10;  ++k) in[din + k] = m[n*10 + k];
  float* t = t0 + (size_t)n*32;
  unsigned short* fa = fA + (size_t)n*32;
  unsigned short* fb = fB + (size_t)n*32;
  for (int j = 0; j < 30; ++j) {
    float s = sB[j];
    for (int k = 0; k < dtot; ++k) s += in[k] * sW[k*30 + j];
    float v = prelu_f(s, a0);
    t[j]  = v;
    fa[j] = f2bf_u(prelu_f(v, a11));
    fb[j] = f2bf_u(prelu_f(v, a12));
  }
  fa[30] = 0; fa[31] = 0; fb[30] = 0; fb[31] = 0;
}

// t1 = prelu(concat(Wa.[t0,aggA,m]+ba, Wb.[t0,aggB,m]+bb), a1)  [N,60]
__global__ void k_combine1(const float* __restrict__ t0,
                           const float* __restrict__ accA, const float* __restrict__ cntA,
                           const float* __restrict__ accB, const float* __restrict__ cntB,
                           const float* __restrict__ m,
                           const float* __restrict__ Wa, const float* __restrict__ Ba,
                           const float* __restrict__ Wb, const float* __restrict__ Bb,
                           const float* __restrict__ a1p, float* __restrict__ t1) {
  __shared__ float sWa[70*30], sBa[30], sWb[70*30], sBb[30];
  for (int i = threadIdx.x; i < 70*30; i += blockDim.x) { sWa[i] = Wa[i]; sWb[i] = Wb[i]; }
  for (int i = threadIdx.x; i < 30;    i += blockDim.x) { sBa[i] = Ba[i]; sBb[i] = Bb[i]; }
  __syncthreads();
  int n = blockIdx.x * blockDim.x + threadIdx.x;
  if (n >= NPROD) return;
  float a1 = a1p[0];
  float in[70];
  const float* t = t0 + (size_t)n*32;
  for (int k = 0; k < 30; ++k) in[k] = t[k];
  float ca = fmaxf(cntA[n], 1.0f);
  for (int k = 0; k < 30; ++k) in[30 + k] = accA[(size_t)n*32 + k] / ca;
  for (int k = 0; k < 10; ++k) in[60 + k] = m[n*10 + k];
  float* o = t1 + (size_t)n*64;
  for (int j = 0; j < 30; ++j) {
    float s = sBa[j];
    for (int k = 0; k < 70; ++k) s += in[k] * sWa[k*30 + j];
    o[j] = prelu_f(s, a1);
  }
  float cb = fmaxf(cntB[n], 1.0f);
  for (int k = 0; k < 30; ++k) in[30 + k] = accB[(size_t)n*32 + k] / cb;
  for (int j = 0; j < 30; ++j) {
    float s = sBb[j];
    for (int k = 0; k < 70; ++k) s += in[k] * sWb[k*30 + j];
    o[30 + j] = prelu_f(s, a1);
  }
}

// featA/B = bf16(prelu(l2_t?_1(t1), a2?))  (60 -> 30)
__global__ void k_feat2(const float* __restrict__ t1,
                        const float* __restrict__ Wa, const float* __restrict__ Ba,
                        const float* __restrict__ a21p,
                        const float* __restrict__ Wb, const float* __restrict__ Bb,
                        const float* __restrict__ a22p,
                        unsigned short* __restrict__ fA, unsigned short* __restrict__ fB) {
  __shared__ float sWa[60*30], sBa[30], sWb[60*30], sBb[30];
  for (int i = threadIdx.x; i < 60*30; i += blockDim.x) { sWa[i] = Wa[i]; sWb[i] = Wb[i]; }
  for (int i = threadIdx.x; i < 30;    i += blockDim.x) { sBa[i] = Ba[i]; sBb[i] = Bb[i]; }
  __syncthreads();
  int n = blockIdx.x * blockDim.x + threadIdx.x;
  if (n >= NPROD) return;
  float a21 = a21p[0], a22 = a22p[0];
  float in[60];
  const float* t = t1 + (size_t)n*64;
  for (int k = 0; k < 60; ++k) in[k] = t[k];
  unsigned short* fa = fA + (size_t)n*32;
  unsigned short* fb = fB + (size_t)n*32;
  for (int j = 0; j < 30; ++j) {
    float sa = sBa[j], sb = sBb[j];
    for (int k = 0; k < 60; ++k) { sa += in[k]*sWa[k*30+j]; sb += in[k]*sWb[k*30+j]; }
    fa[j] = f2bf_u(prelu_f(sa, a21));
    fb[j] = f2bf_u(prelu_f(sb, a22));
  }
  fa[30] = 0; fa[31] = 0; fb[30] = 0; fb[31] = 0;
}

// h = prelu(concat(Wa.[t1,aggA,m]+ba (15), Wb.[t1,aggB,m]+bb (15)), a2) [N,30]
__global__ void k_combine2(const float* __restrict__ t1,
                           const float* __restrict__ accA, const float* __restrict__ cntA,
                           const float* __restrict__ accB, const float* __restrict__ cntB,
                           const float* __restrict__ m,
                           const float* __restrict__ Wa, const float* __restrict__ Ba,
                           const float* __restrict__ Wb, const float* __restrict__ Bb,
                           const float* __restrict__ a2p, float* __restrict__ hout) {
  __shared__ float sWa[100*15], sBa[15], sWb[100*15], sBb[15];
  for (int i = threadIdx.x; i < 100*15; i += blockDim.x) { sWa[i] = Wa[i]; sWb[i] = Wb[i]; }
  for (int i = threadIdx.x; i < 15;     i += blockDim.x) { sBa[i] = Ba[i]; sBb[i] = Bb[i]; }
  __syncthreads();
  int n = blockIdx.x * blockDim.x + threadIdx.x;
  if (n >= NPROD) return;
  float a2 = a2p[0];
  float in[100];
  const float* t = t1 + (size_t)n*64;
  for (int k = 0; k < 60; ++k) in[k] = t[k];
  float ca = fmaxf(cntA[n], 1.0f);
  for (int k = 0; k < 30; ++k) in[60 + k] = accA[(size_t)n*32 + k] / ca;
  for (int k = 0; k < 10; ++k) in[90 + k] = m[n*10 + k];
  float* o = hout + (size_t)n*32;
  for (int j = 0; j < 15; ++j) {
    float s = sBa[j];
    for (int k = 0; k < 100; ++k) s += in[k] * sWa[k*15 + j];
    o[j] = prelu_f(s, a2);
  }
  float cb = fmaxf(cntB[n], 1.0f);
  for (int k = 0; k < 30; ++k) in[60 + k] = accB[(size_t)n*32 + k] / cb;
  for (int j = 0; j < 15; ++j) {
    float s = sBb[j];
    for (int k = 0; k < 100; ++k) s += in[k] * sWb[k*15 + j];
    o[15 + j] = prelu_f(s, a2);
  }
}

// Bipartite readout message pass: per product node -> atomic scatter to M tgts
__global__ void k_bip(const float* __restrict__ h, const float* __restrict__ m,
                      const int* __restrict__ tgt,
                      const f32x4* __restrict__ posP, const f32x4* __restrict__ posM,
                      const float* __restrict__ W1a, const float* __restrict__ B1a,
                      const float* __restrict__ W1b, const float* __restrict__ B1b,
                      const float* __restrict__ ainp, const float* __restrict__ a1p,
                      float* __restrict__ acc, float* __restrict__ cnt) {
  __shared__ float sWa[43*30], sBa[30], sWb[30*30], sBb[30];
  for (int i = threadIdx.x; i < 43*30; i += blockDim.x) sWa[i] = W1a[i];
  for (int i = threadIdx.x; i < 30*30; i += blockDim.x) sWb[i] = W1b[i];
  for (int i = threadIdx.x; i < 30;    i += blockDim.x) { sBa[i] = B1a[i]; sBb[i] = B1b[i]; }
  __syncthreads();
  int n = blockIdx.x * blockDim.x + threadIdx.x;
  if (n >= NPROD) return;
  float ain = ainp[0], a1 = a1p[0];
  float in[43];
  const float* hr = h + (size_t)n*32;
  for (int k = 0; k < 30; ++k) in[k] = hr[k];
  for (int k = 0; k < 10; ++k) in[30 + k] = m[n*10 + k];
  f32x4 pp = posP[n], pm = posM[n];
  in[40] = (pp.x - pm.x) * INV_SCALE;
  in[41] = (pp.y - pm.y) * INV_SCALE;
  in[42] = (pp.z - pm.z) * INV_SCALE;
  float h1[30];
  for (int j = 0; j < 30; ++j) {
    float s = sBa[j];
    for (int k = 0; k < 43; ++k) s += in[k] * sWa[k*30 + j];
    h1[j] = prelu_f(s, ain);
  }
  int t = tgt[n];
  float* arow = acc + (size_t)t*32;
  for (int j = 0; j < 30; ++j) {
    float s = sBb[j];
    for (int k = 0; k < 30; ++k) s += h1[k] * sWb[k*30 + j];
    atomicAdd(arow + j, prelu_f(s, a1));
  }
  atomicAdd(cnt + t, 1.0f);
}

__global__ void k_bipfin(const float* __restrict__ acc, const float* __restrict__ cnt,
                         const float* __restrict__ W2, const float* __restrict__ B2,
                         const float* __restrict__ a2p, float* __restrict__ x, int M) {
  int i = blockIdx.x * blockDim.x + threadIdx.x;
  if (i >= M) return;
  float c = fmaxf(cnt[i], 1.0f), a2 = a2p[0];
  float ag[30];
  for (int k = 0; k < 30; ++k) ag[k] = acc[(size_t)i*32 + k] / c;
  for (int j = 0; j < 15; ++j) {
    float s = B2[j];
    for (int k = 0; k < 30; ++k) s += ag[k] * W2[k*15 + j];
    x[i*16 + j] = prelu_f(s, a2);
  }
}

__device__ void mlp2g(const float* in, int d0, int d1, int d2,
                      const float* W0, const float* B0, float a,
                      const float* W1, const float* B1, float scale, float* out) {
  float hb[32];
  for (int j = 0; j < d1; ++j) {
    float s = B0[j];
    for (int k = 0; k < d0; ++k) s += in[k] * W0[k*d1 + j];
    hb[j] = prelu_f(s, a);
  }
  for (int j = 0; j < d2; ++j) {
    float s = B1[j];
    for (int k = 0; k < d1; ++k) s += hb[k] * W1[k*d2 + j];
    out[j] = s * scale;
  }
}

__global__ void k_final(const float* __restrict__ x1, const float* __restrict__ x2,
                        const float* pW0, const float* pB0, const float* pA,
                        const float* pW1, const float* pB1,
                        const float* tW0, const float* tB0, const float* tA,
                        const float* tW1, const float* tB1,
                        const float* cW0, const float* cB0, const float* cAp,
                        const float* cW1, const float* cB1,
                        float* __restrict__ out) {
  int i = blockIdx.x * blockDim.x + threadIdx.x;
  if (i < NSRC) {
    float in[15];
    for (int k = 0; k < 15; ++k) in[k] = x1[i*16 + k];
    float o3[3];
    mlp2g(in, 15, 30, 3, pW0, pB0, pA[0], pW1, pB1, 5000.0f, o3);
    out[i*3 + 0] = o3[0]; out[i*3 + 1] = o3[1]; out[i*3 + 2] = o3[2];
    float o1[1];
    mlp2g(in, 15, 15, 1, tW0, tB0, tA[0], tW1, tB1, 1.0f, o1);
    out[1500 + i] = o1[0];
  }
  if (i < NSTA) {
    float in[15];
    for (int k = 0; k < 15; ++k) in[k] = x2[i*16 + k];
    float o2[2];
    mlp2g(in, 15, 15, 2, cW0, cB0, cAp[0], cW1, cB1, 1.0f, o2);
    out[2000 + i*2 + 0] = o2[0]; out[2000 + i*2 + 1] = o2[1];
  }
}

__global__ void k_copyh(const float* __restrict__ h, float* __restrict__ out) {
  long idx = (long)blockIdx.x * blockDim.x + threadIdx.x;
  if (idx >= (long)NPROD * 30) return;
  long n = idx / 30, j = idx % 30;
  out[idx] = h[(size_t)n*32 + j];
}

// ---------------------------------------------------------------------------
extern "C" void kernel_launch(void* const* d_in, const int* in_sizes, int n_in,
                              void* d_out, int out_size, void* d_ws, size_t ws_size,
                              hipStream_t stream) {
  (void)in_sizes; (void)n_in; (void)out_size; (void)ws_size;
  const float* x    = (const float*)d_in[0];
  const float* mask = (const float*)d_in[1];
  const int* pick   = (const int*)d_in[2];
  const int* insrc  = (const int*)d_in[3];
  const int* ss     = (const int*)d_in[6];
  const float* locs = (const float*)d_in[7];
  const float* srcs = (const float*)d_in[8];
  const int* sta_idx = ss;
  const int* src_idx = ss + NPROD;
  auto P = [&](int leaf) -> const float* { return (const float*)d_in[9 + leaf]; };
  // leaf bases: bip1=0, bip2=9, daL=18+24*L, embed=138, proj=143, proj_c=148, proj_t=153

  float* ws = (float*)d_ws;
  size_t off = 0;
  auto alloc = [&](size_t nf) -> size_t { size_t r = off; off += (nf + 63) & ~(size_t)63; return r; };
  size_t o_m    = alloc((size_t)NPROD * 10);
  size_t o_t0   = alloc((size_t)NPROD * 32);
  size_t o_t1   = alloc((size_t)NPROD * 64);
  size_t o_h    = alloc((size_t)NPROD * 32);
  size_t o_fA   = alloc((size_t)NPROD * 16);   // bf16[N,32]
  size_t o_fB   = alloc((size_t)NPROD * 16);
  size_t o_accA = alloc((size_t)NPROD * 32);
  size_t o_accB = alloc((size_t)NPROD * 32);
  size_t o_cA   = alloc((size_t)NPROD);
  size_t o_cB   = alloc((size_t)NPROD);
  size_t accZeroEnd = o_cB + (size_t)NPROD;
  size_t o_pS   = alloc((size_t)NPROD * 4);
  size_t o_pR   = alloc((size_t)NPROD * 4);
  size_t o_x1a  = alloc((size_t)NSRC * 32);
  size_t o_x1c  = alloc((size_t)NSRC);
  size_t o_x2a  = alloc((size_t)NSTA * 32);
  size_t o_x2c  = alloc((size_t)NSTA);
  size_t bipZeroEnd = o_x2c + (size_t)NSTA;
  size_t o_x1   = alloc((size_t)NSRC * 16);
  size_t o_x2   = alloc((size_t)NSTA * 16);

  float* m_   = ws + o_m;
  float* t0   = ws + o_t0;
  float* t1   = ws + o_t1;
  float* h    = ws + o_h;
  unsigned short* fA = (unsigned short*)(ws + o_fA);
  unsigned short* fB = (unsigned short*)(ws + o_fB);
  float* accA = ws + o_accA; float* accB = ws + o_accB;
  float* cA   = ws + o_cA;   float* cB   = ws + o_cB;
  f32x4* posS = (f32x4*)(ws + o_pS);
  f32x4* posR = (f32x4*)(ws + o_pR);
  float* x1a = ws + o_x1a; float* x1c = ws + o_x1c;
  float* x2a = ws + o_x2a; float* x2c = ws + o_x2c;
  float* x1  = ws + o_x1;  float* x2  = ws + o_x2;

  const int NB = (NPROD + 255) / 256;
  const int NT = NEDGE / 16;

  k_nodepos<<<NB, 256, 0, stream>>>(sta_idx, src_idx, locs, srcs, posS, posR);
  k_embed<<<NB, 256, 0, stream>>>(mask, P(140), P(139), P(138), P(142), P(141), m_);

  const float* hin = x;
  int hstride = 18, din = 18;
  for (int L = 0; L < 5; ++L) {
    int b = 18 + 24 * L;
    k_init<<<NB, 256, 0, stream>>>(hin, hstride, din, m_, P(b+9), P(b+8),
                                   P(b+0), P(b+2), P(b+3), t0, fA, fB);
    hipMemsetAsync(ws + o_accA, 0, (accZeroEnd - o_accA) * sizeof(float), stream);
    k_edge_wmma<<<1024, 256, 0, stream>>>(fA, pick, pick + NEDGE, posS,
                                          P(b+23), P(b+22), P(b+7), accA, cA, NT);
    k_edge_wmma<<<1024, 256, 0, stream>>>(fB, insrc, insrc + NEDGE, posR,
                                          P(b+23), P(b+22), P(b+7), accB, cB, NT);
    k_combine1<<<NB, 256, 0, stream>>>(t0, accA, cA, accB, cB, m_,
                                       P(b+11), P(b+10), P(b+13), P(b+12), P(b+1), t1);
    k_feat2<<<NB, 256, 0, stream>>>(t1, P(b+15), P(b+14), P(b+5),
                                    P(b+19), P(b+18), P(b+6), fA, fB);
    hipMemsetAsync(ws + o_accA, 0, (accZeroEnd - o_accA) * sizeof(float), stream);
    k_edge_wmma<<<1024, 256, 0, stream>>>(fA, pick, pick + NEDGE, posS,
                                          P(b+23), P(b+22), P(b+7), accA, cA, NT);
    k_edge_wmma<<<1024, 256, 0, stream>>>(fB, insrc, insrc + NEDGE, posR,
                                          P(b+23), P(b+22), P(b+7), accB, cB, NT);
    k_combine2<<<NB, 256, 0, stream>>>(t1, accA, cA, accB, cB, m_,
                                       P(b+17), P(b+16), P(b+21), P(b+20), P(b+4), h);
    hin = h; hstride = 32; din = 30;
  }

  hipMemsetAsync(ws + o_x1a, 0, (bipZeroEnd - o_x1a) * sizeof(float), stream);
  // bip1: rel = (srcs[src_idx] - locs[sta_idx])/30000, targets = src_idx (M=500)
  k_bip<<<NB, 256, 0, stream>>>(h, m_, src_idx, posR, posS,
                                P(4), P(3), P(6), P(5), P(2), P(0), x1a, x1c);
  // bip2: rel = (locs[sta_idx] - srcs[src_idx])/30000, targets = sta_idx (M=200)
  k_bip<<<NB, 256, 0, stream>>>(h, m_, sta_idx, posS, posR,
                                P(13), P(12), P(15), P(14), P(11), P(9), x2a, x2c);
  k_bipfin<<<2, 256, 0, stream>>>(x1a, x1c, P(8),  P(7),  P(1),  x1, NSRC);
  k_bipfin<<<1, 256, 0, stream>>>(x2a, x2c, P(17), P(16), P(10), x2, NSTA);

  k_final<<<1, 512, 0, stream>>>(x1, x2,
      P(145), P(144), P(143), P(147), P(146),   // proj
      P(155), P(154), P(153), P(157), P(156),   // proj_t
      P(150), P(149), P(148), P(152), P(151),   // proj_c
      (float*)d_out);
  long tot = (long)NPROD * 30;
  k_copyh<<<(int)((tot + 255) / 256), 256, 0, stream>>>(h, (float*)d_out + 2400);
}
